// Attention_53480932770284
// MI455X (gfx1250) — matrix operand order repaired and checked by reading, not compiled, not run on hardware
//
#include <hip/hip_runtime.h>
#include <hip/hip_bf16.h>
#include <math.h>

// ---------------------------------------------------------------------------
// Types for CDNA5 WMMA (wave32): v_wmma_f32_16x16x32_bf16
// ---------------------------------------------------------------------------
typedef __attribute__((ext_vector_type(8)))  float        v8f;
typedef __attribute__((ext_vector_type(16))) __bf16       v16bf;
typedef __attribute__((ext_vector_type(4)))  unsigned int u32x4;
typedef __attribute__((ext_vector_type(4)))  float        f32x4;

union Frag16 {
    v16bf bf;
    u32x4 q[2];
};

__device__ __forceinline__ unsigned short f32_bf16(float f) {
    union { float f; unsigned u; } v; v.f = f;
    unsigned r = v.u + 0x7FFFu + ((v.u >> 16) & 1u);   // round-to-nearest-even
    return (unsigned short)(r >> 16);
}
__device__ __forceinline__ unsigned pack_bf16x2(float lo, float hi) {
    return (unsigned)f32_bf16(lo) | ((unsigned)f32_bf16(hi) << 16);
}

__device__ __forceinline__ v8f wmma_bf16(const Frag16& a, const Frag16& b, v8f c) {
    return __builtin_amdgcn_wmma_f32_16x16x32_bf16(
        false, a.bf, false, b.bf, (short)0, c, false, false);
}

// Async global->LDS copy (CDNA5, ASYNCcnt). LDS operand = byte offset within
// the workgroup LDS allocation (low 32 bits of the flat shared address).
__device__ __forceinline__ void async_load_b128(unsigned lds_byte, const void* gptr) {
    asm volatile("global_load_async_to_lds_b128 %0, %1, off"
                 :: "v"(lds_byte), "v"(gptr) : "memory");
}
__device__ __forceinline__ void wait_asynccnt0() {
    asm volatile("s_wait_asynccnt 0x0" ::: "memory");
}
__device__ __forceinline__ unsigned lds_off(const void* p) {
    return (unsigned)(unsigned long long)p;
}

// Problem constants
#define BB   2
#define NN   2048
#define DD   1024
#define HH   16
#define DHH  64

// ---------------------------------------------------------------------------
// Elementwise f32 -> bf16 conversion (8 elems / thread, b128 in / b128 out)
// ---------------------------------------------------------------------------
__global__ __launch_bounds__(256) void convert_bf16_kernel(
    const float* __restrict__ x, unsigned short* __restrict__ y)
{
    size_t i = ((size_t)blockIdx.x * 256 + threadIdx.x) * 8;
    f32x4 a0 = *(const f32x4*)(x + i);
    f32x4 a1 = *(const f32x4*)(x + i + 4);
    u32x4 pk;
    pk[0] = pack_bf16x2(a0[0], a0[1]);
    pk[1] = pack_bf16x2(a0[2], a0[3]);
    pk[2] = pack_bf16x2(a1[0], a1[1]);
    pk[3] = pack_bf16x2(a1[2], a1[3]);
    *(u32x4*)(y + i) = pk;
}

// ---------------------------------------------------------------------------
// Tiled transpose + convert: W [K][N] f32 (row-major)  ->  WT [N][K] bf16
// 64x64 tiles through LDS; coalesced reads and b128 writes.
// ---------------------------------------------------------------------------
__global__ __launch_bounds__(256) void transpose_bf16_kernel(
    const float* __restrict__ W, unsigned short* __restrict__ WT, int K, int N)
{
    __shared__ __align__(16) unsigned short t[64 * 80];   // pad stride 80
    const int tid = threadIdx.x;
    const int k0 = blockIdx.x * 64, n0 = blockIdx.y * 64;
#pragma unroll
    for (int i = 0; i < 16; ++i) {
        int idx = tid + i * 256;
        int kk = idx >> 6, nn = idx & 63;
        t[nn * 80 + kk] = f32_bf16(W[(size_t)(k0 + kk) * N + (n0 + nn)]);
    }
    __syncthreads();
#pragma unroll
    for (int i = 0; i < 2; ++i) {
        int chunk = tid + i * 256;            // 512 chunks of 8 halves
        int nn = chunk >> 3, k8 = (chunk & 7) * 8;
        *(u32x4*)&WT[(size_t)(n0 + nn) * K + (k0 + k8)] = *(const u32x4*)&t[nn * 80 + k8];
    }
}

// ---------------------------------------------------------------------------
// Pure-bf16 GEMM: C[f32] = A[bf16, MxK] @ BT[bf16, NxK]^T
// Block 128x128, 8 waves of 64x32. Double-buffered LDS, staged entirely with
// async global->LDS b128 copies (no conversion VALU in the loop).
// ---------------------------------------------------------------------------
__device__ __forceinline__ void gemm_stage(
    unsigned short* lA, unsigned short* lB,
    const unsigned short* Ab, const unsigned short* Bt,
    int tid, int bm, int bn, int K, int k0)
{
#pragma unroll
    for (int i = 0; i < 2; ++i) {
        int t = tid + i * 256;                // 512 x b128 per tile
        int m = t >> 2, c8 = (t & 3) * 8;
        async_load_b128(lds_off(lA + m * 32 + c8), Ab + (size_t)(bm + m) * K + k0 + c8);
        async_load_b128(lds_off(lB + m * 32 + c8), Bt + (size_t)(bn + m) * K + k0 + c8);
    }
}

__global__ __launch_bounds__(256) void gemm_async_kernel(
    const unsigned short* __restrict__ Ab,   // [M][K] bf16
    const unsigned short* __restrict__ Bt,   // [N][K] bf16 (pre-transposed)
    float* __restrict__ C, int K, int ldc)
{
    __shared__ __align__(16) unsigned short lA[2][128 * 32];
    __shared__ __align__(16) unsigned short lB[2][128 * 32];

    const int tid  = threadIdx.x;
    const int lane = tid & 31;
    const int wid  = tid >> 5;
    const int half = lane >> 4;
    const int l16  = lane & 15;
    const int bm = blockIdx.x * 128;
    const int bn = blockIdx.y * 128;
    const int wm = (wid >> 2) * 64;
    const int wn = (wid & 3) * 32;

    v8f acc[4][2];
#pragma unroll
    for (int i = 0; i < 4; ++i)
#pragma unroll
        for (int j = 0; j < 2; ++j)
#pragma unroll
            for (int e = 0; e < 8; ++e) acc[i][j][e] = 0.f;

    gemm_stage(lA[0], lB[0], Ab, Bt, tid, bm, bn, K, 0);

    int cur = 0;
    for (int k0 = 0; k0 < K; k0 += 32) {
        wait_asynccnt0();
        __syncthreads();
        if (k0 + 32 < K)
            gemm_stage(lA[cur ^ 1], lB[cur ^ 1], Ab, Bt, tid, bm, bn, K, k0 + 32);

        Frag16 af[4], bf[2];
        const int ka = half * 8;
#pragma unroll
        for (int mt = 0; mt < 4; ++mt) {
            const unsigned short* p = &lA[cur][(wm + mt * 16 + l16) * 32];
            af[mt].q[0] = *(const u32x4*)(p + ka);
            af[mt].q[1] = *(const u32x4*)(p + 16 + ka);
        }
        const int kb = half * 8;
#pragma unroll
        for (int nt = 0; nt < 2; ++nt) {
            const unsigned short* p = &lB[cur][(wn + nt * 16 + l16) * 32];
            bf[nt].q[0] = *(const u32x4*)(p + kb);
            bf[nt].q[1] = *(const u32x4*)(p + 16 + kb);
        }
#pragma unroll
        for (int mt = 0; mt < 4; ++mt)
#pragma unroll
            for (int nt = 0; nt < 2; ++nt)
                acc[mt][nt] = wmma_bf16(af[mt], bf[nt], acc[mt][nt]);
        cur ^= 1;
    }

#pragma unroll
    for (int mt = 0; mt < 4; ++mt)
#pragma unroll
        for (int nt = 0; nt < 2; ++nt)
#pragma unroll
            for (int r = 0; r < 8; ++r) {
                int row = bm + wm + mt * 16 + r + 8 * half;
                int col = bn + wn + nt * 16 + l16;
                C[(size_t)row * ldc + col] = acc[mt][nt][r];
            }
}

// ---------------------------------------------------------------------------
// Epilogue: bq/bg bias, RoPE on q/k, q-scale, sigmoid gate; emit bf16
// q/k/v in [b][h][n][64] layout.
// ---------------------------------------------------------------------------
__global__ __launch_bounds__(256) void epilogue_kernel(
    const float* __restrict__ qkvg, const float* __restrict__ bq,
    const float* __restrict__ bg,
    unsigned short* __restrict__ qb, unsigned short* __restrict__ kb,
    unsigned short* __restrict__ vb, float* __restrict__ gates)
{
    int idx = blockIdx.x * 256 + threadIdx.x;    // [0, 4096*1024)
    int row = idx >> 10;
    int c   = idx & 1023;
    int h = c >> 6, d = c & 63;
    int b = row >> 11, pos = row & 2047;
    size_t base = (size_t)row * 4096;

    int cp = (d < 32) ? c + 32 : c - 32;         // rotate_half partner
    float q  = qkvg[base + c]         + bq[c];
    float qp = qkvg[base + cp]        + bq[cp];
    float k  = qkvg[base + 1024 + c];
    float kp = qkvg[base + 1024 + cp];
    float v  = qkvg[base + 2048 + c];
    float g  = qkvg[base + 3072 + c]  + bg[c];

    int dr = d & 31;
    float inv = __powf(1024.0f, -(float)dr * (1.0f / 32.0f));  // THETA^(-2i/DH)
    float ang = (float)pos * inv;
    float cs = __cosf(ang), sn = __sinf(ang);
    float rq = (d < 32) ? -qp : qp;
    float rk = (d < 32) ? -kp : kp;
    float qo = (q * cs + rq * sn) * 0.125f;      // * DH^-0.5
    float ko =  k * cs + rk * sn;

    size_t o = (((size_t)b * HH + h) * NN + pos) * DHH + d;
    qb[o] = f32_bf16(qo);
    kb[o] = f32_bf16(ko);
    vb[o] = f32_bf16(v);
    gates[(size_t)row * 1024 + c] = 1.0f / (1.0f + __expf(-g));
}

// ---------------------------------------------------------------------------
// Flash attention with static softmax max (softclamp bounds scores to +-50,
// so exp(s-50) is safe): no online max, no shuffle reductions. Row sums are
// produced by an extra WMMA against an all-ones B tile (same C-layout as O).
// Double-buffered K (async b128 copies) / V (transposing copy). Applies the
// sigmoid gate at the store and emits bf16 for the final GEMM.
// ---------------------------------------------------------------------------
__global__ __launch_bounds__(256) void attn_kernel(
    const unsigned short* __restrict__ qb,
    const unsigned short* __restrict__ kbp,
    const unsigned short* __restrict__ vbp,
    const float* __restrict__ bias,           // [B,1,N,N]
    const unsigned char* __restrict__ mask,   // [B,N]
    const float* __restrict__ gates,          // [B*N][1024]
    unsigned short* __restrict__ gatedb)      // [B*N][1024] bf16
{
    __shared__ __align__(16) unsigned short sK[2][64 * 64];    // [key][d]
    __shared__ __align__(16) unsigned short sVT[2][64 * 64];   // [d][key]
    __shared__ __align__(16) unsigned short sP[8][16 * 64];    // per-wave [m][key]

    const int tid  = threadIdx.x;
    const int lane = tid & 31;
    const int w    = tid >> 5;
    const int half = lane >> 4;
    const int l16  = lane & 15;
    const int bh = blockIdx.x;                 // b*16 + h
    const int b  = bh >> 4;
    const int qbase = blockIdx.y * 128 + w * 16;
    const int ka = half * 8;
    const int kb16 = half * 8;

    // Q fragments for this wave's 16 rows (two K-steps over d)
    Frag16 qf[2];
    {
        const unsigned short* qr = qb + ((size_t)bh * NN + (size_t)(qbase + l16)) * DHH;
        qf[0].q[0] = *(const u32x4*)(qr + ka);
        qf[0].q[1] = *(const u32x4*)(qr + 16 + ka);
        qf[1].q[0] = *(const u32x4*)(qr + 32 + ka);
        qf[1].q[1] = *(const u32x4*)(qr + 48 + ka);
    }

    // all-ones bf16 B fragment (for row-sum WMMA)
    Frag16 ones;
#pragma unroll
    for (int e = 0; e < 4; ++e) { ones.q[0][e] = 0x3F803F80u; ones.q[1][e] = 0x3F803F80u; }

    v8f oacc[4], lsum;
#pragma unroll
    for (int dt = 0; dt < 4; ++dt)
#pragma unroll
        for (int e = 0; e < 8; ++e) oacc[dt][e] = 0.f;
#pragma unroll
    for (int e = 0; e < 8; ++e) lsum[e] = 0.f;

    // ---- stage helper (K async, V transposing copy) ----
    auto stageKV = [&](int buf, int kt) {
        const size_t kvbase = ((size_t)bh * NN + kt) * DHH;
#pragma unroll
        for (int i = 0; i < 2; ++i) {
            int t = tid + i * 256;
            async_load_b128(lds_off(&sK[buf][t * 8]), kbp + kvbase + t * 8);
        }
#pragma unroll
        for (int i = 0; i < 16; ++i) {
            int idx = tid + i * 256;
            int key = idx >> 6, dd = idx & 63;
            sVT[buf][dd * 64 + key] = vbp[kvbase + idx];
        }
    };

    stageKV(0, 0);
    wait_asynccnt0();
    __syncthreads();

    int cur = 0;
    for (int kt = 0; kt < NN; kt += 64) {
        if (kt + 64 < NN) stageKV(cur ^ 1, kt + 64);

        // ---- S = q @ k^T : 4 key-subtiles x 2 d-steps -> 8 WMMA ----
        v8f s[4];
#pragma unroll
        for (int j = 0; j < 4; ++j) {
#pragma unroll
            for (int e = 0; e < 8; ++e) s[j][e] = 0.f;
            const unsigned short* kp = &sK[cur][(j * 16 + l16) * 64];
            Frag16 b0, b1;
            b0.q[0] = *(const u32x4*)(kp + kb16);
            b0.q[1] = *(const u32x4*)(kp + 16 + kb16);
            b1.q[0] = *(const u32x4*)(kp + 32 + kb16);
            b1.q[1] = *(const u32x4*)(kp + 48 + kb16);
            s[j] = wmma_bf16(qf[0], b0, s[j]);
            s[j] = wmma_bf16(qf[1], b1, s[j]);
        }

        // ---- bias (NT stream) + softclamp + mask + exp(s-50) -> P bf16 ----
#pragma unroll
        for (int j = 0; j < 4; ++j) {
            int key = kt + j * 16 + l16;
            unsigned char mk = mask[b * NN + key];
#pragma unroll
            for (int r = 0; r < 8; ++r) {
                int qrow = qbase + r + 8 * half;
                float x = s[j][r];
                x += __builtin_nontemporal_load(&bias[((size_t)b * NN + qrow) * NN + key]);
                // 50*tanh(x/50) via one exp: tanh(t)=sign(t)*(1-e)/(1+e), e=exp(-2|t|)
                float t  = x * 0.02f;
                float e2 = __expf(-2.0f * fabsf(t));
                float th = (1.0f - e2) * __builtin_amdgcn_rcpf(1.0f + e2);
                float y  = 50.0f * copysignf(th, t);
                float p  = __expf(y - 50.0f);          // static max = clamp bound
                if (!mk) p = 0.f;
                sP[w][(r + 8 * half) * 64 + j * 16 + l16] = f32_bf16(p);
            }
        }

        // ---- O += P @ V, lsum += P @ ones : 10 WMMA ----
        Frag16 pa0, pa1;
        {
            const unsigned short* pr = &sP[w][l16 * 64];
            pa0.q[0] = *(const u32x4*)(pr + ka);
            pa0.q[1] = *(const u32x4*)(pr + 16 + ka);
            pa1.q[0] = *(const u32x4*)(pr + 32 + ka);
            pa1.q[1] = *(const u32x4*)(pr + 48 + ka);
        }
        lsum = wmma_bf16(pa0, ones, lsum);
        lsum = wmma_bf16(pa1, ones, lsum);
#pragma unroll
        for (int dt = 0; dt < 4; ++dt) {
            const unsigned short* vp = &sVT[cur][(dt * 16 + l16) * 64];
            Frag16 vb0, vb1;
            vb0.q[0] = *(const u32x4*)(vp + kb16);
            vb0.q[1] = *(const u32x4*)(vp + 16 + kb16);
            vb1.q[0] = *(const u32x4*)(vp + 32 + kb16);
            vb1.q[1] = *(const u32x4*)(vp + 48 + kb16);
            oacc[dt] = wmma_bf16(pa0, vb0, oacc[dt]);
            oacc[dt] = wmma_bf16(pa1, vb1, oacc[dt]);
        }

        wait_asynccnt0();     // next buffer's async copies done
        __syncthreads();      // everyone finished reading current buffer
        cur ^= 1;
    }

    // ---- finalize: normalize, gate, emit bf16 (merged heads) ----
    const int h = bh & 15;
#pragma unroll
    for (int r = 0; r < 8; ++r) {
        float invl = __builtin_amdgcn_rcpf(lsum[r]);
        int qrow = qbase + r + 8 * half;
        size_t rowbase = ((size_t)b * NN + qrow) * (HH * DHH) + h * DHH;
#pragma unroll
        for (int dt = 0; dt < 4; ++dt) {
            int dcol = dt * 16 + l16;
            float gv = gates[rowbase + dcol];
            gatedb[rowbase + dcol] = f32_bf16(oacc[dt][r] * invl * gv);
        }
    }
}

// ---------------------------------------------------------------------------
// Host-side launcher
// ---------------------------------------------------------------------------
extern "C" void kernel_launch(void* const* d_in, const int* in_sizes, int n_in,
                              void* d_out, int out_size, void* d_ws, size_t ws_size,
                              hipStream_t stream) {
    const float*         seq  = (const float*)d_in[0];
    const unsigned char* mask = (const unsigned char*)d_in[1];
    const float*         bias = (const float*)d_in[2];
    const float*         Wq   = (const float*)d_in[3];
    const float*         bq   = (const float*)d_in[4];
    const float*         Wkv  = (const float*)d_in[5];
    const float*         Wg   = (const float*)d_in[6];
    const float*         bg   = (const float*)d_in[7];
    const float*         Wo   = (const float*)d_in[8];
    float* out = (float*)d_out;

    char* ws = (char*)d_ws;
    const size_t MB = 1024u * 1024u;
    float*          qkvg   = (float*)(ws);                      // 64 MB [4096][4096]
    unsigned short* qb     = (unsigned short*)(ws + 64 * MB);   //  8 MB [b][h][n][64]
    unsigned short* kb     = (unsigned short*)(ws + 72 * MB);   //  8 MB
    unsigned short* vb     = (unsigned short*)(ws + 80 * MB);   //  8 MB
    float*          gates  = (float*)(ws + 88 * MB);            // 16 MB [4096][1024]
    unsigned short* seqb   = (unsigned short*)(ws + 104 * MB);  //  8 MB [4096][1024]
    unsigned short* WqT    = (unsigned short*)(ws + 112 * MB);  //  2 MB [1024][1024]
    unsigned short* WkvT   = (unsigned short*)(ws + 114 * MB);  //  4 MB [2048][1024]
    unsigned short* WgT    = (unsigned short*)(ws + 118 * MB);  //  2 MB
    unsigned short* WoT    = (unsigned short*)(ws + 120 * MB);  //  2 MB
    unsigned short* gatedb = (unsigned short*)(ws + 122 * MB);  //  8 MB [4096][1024]

    dim3 blk(256);

    // One-time conversions / transposes
    convert_bf16_kernel<<<dim3(2048), blk, 0, stream>>>(seq, seqb);            // 4096x1024
    transpose_bf16_kernel<<<dim3(16, 16), blk, 0, stream>>>(Wq,  WqT,  1024, 1024);
    transpose_bf16_kernel<<<dim3(16, 32), blk, 0, stream>>>(Wkv, WkvT, 1024, 2048);
    transpose_bf16_kernel<<<dim3(16, 16), blk, 0, stream>>>(Wg,  WgT,  1024, 1024);
    transpose_bf16_kernel<<<dim3(16, 16), blk, 0, stream>>>(Wo,  WoT,  1024, 1024);

    // Projections into combined [4096][4096] f32 buffer (ldc = 4096)
    gemm_async_kernel<<<dim3(32, 8),  blk, 0, stream>>>(seqb, WqT,  qkvg,        1024, 4096);
    gemm_async_kernel<<<dim3(32, 16), blk, 0, stream>>>(seqb, WkvT, qkvg + 1024, 1024, 4096);
    gemm_async_kernel<<<dim3(32, 8),  blk, 0, stream>>>(seqb, WgT,  qkvg + 3072, 1024, 4096);

    // RoPE / bias / sigmoid epilogue
    epilogue_kernel<<<dim3(16384), blk, 0, stream>>>(qkvg, bq, bg, qb, kb, vb, gates);

    // Flash attention (writes gated bf16 output)
    attn_kernel<<<dim3(32, 16), blk, 0, stream>>>(qb, kb, vb, bias, mask, gates, gatedb);

    // Output projection: out = gated @ Wo
    gemm_async_kernel<<<dim3(32, 8), blk, 0, stream>>>(gatedb, WoT, out, 1024, 1024);
}
